// EGNNEquiHNNS_84155589198112
// MI455X (gfx1250) — compile-verified, matrix operand order checked by reading, not run on hardware
//
#include <hip/hip_runtime.h>

typedef unsigned short u16;
typedef unsigned int   u32;

typedef __attribute__((ext_vector_type(16))) __bf16 v16bf;
typedef __attribute__((ext_vector_type(8)))  float  v8f;
typedef __attribute__((ext_vector_type(8)))  unsigned int v8u;
typedef __attribute__((ext_vector_type(4)))  unsigned int v4u;

union Frag { v16bf bf; v8u u; struct { v4u lo, hi; } h; };

#define Nn   8192
#define DIMC 256
#define KNNc 16
#define NEc  16384
#define NNZc 65536
#define Bc   512
#define EK   544    // edge MLP input dim 513 -> padded to 17*32
#define EH   1056   // edge MLP hidden dim 1026 -> padded to 33*32

__device__ __forceinline__ u16 f2bf(float f) {
  u32 u = __float_as_uint(f);
  u32 r = (u + 0x7FFFu + ((u >> 16) & 1u)) >> 16;   // RNE
  return (u16)r;
}
__device__ __forceinline__ float bf2f(u16 h) {
  return __uint_as_float(((u32)h) << 16);
}
// K-offset pattern for 16-bit A/B fragments (ISA 7.12.2, 16-bit A 16x32)
__device__ __forceinline__ int kfrag(int p, int hh) {
  return ((p < 4) ? (p << 1) : (16 + ((p - 4) << 1))) + (hh << 3);
}
__device__ __forceinline__ float silu_f(float v) { return v / (1.0f + __expf(-v)); }

// ---------------------------------------------------------------------------
// Weight pre-pack: f32 [rows, cols] -> bf16 fragment-major, zero-padded to
// [Kd, Nd]. One 32-lane block per (n-tile, k-step); each lane owns the 8
// dword-pairs of its B fragment, stored contiguously (32B per lane).
// ---------------------------------------------------------------------------
__global__ __launch_bounds__(32)
void pack_w_kernel(const float* __restrict__ src, int rows, int cols,
                   u32* __restrict__ out, int Kd, int Nd) {
  int nk = Kd >> 5;
  int tile = blockIdx.x;               // tn * nk + kbi
  int tn = tile / nk, kbi = tile % nk;
  int lane = threadIdx.x;
  int m = lane & 15, hh = lane >> 4;
  int col = (tn << 4) + m;
  u32* dst = out + ((size_t)tile * 32 + lane) * 8;
#pragma unroll
  for (int p = 0; p < 8; ++p) {
    int k0 = (kbi << 5) + kfrag(p, hh);
    float f0 = (k0 < rows && col < cols) ? src[(size_t)k0 * cols + col] : 0.0f;
    float f1 = (k0 + 1 < rows && col < cols) ? src[(size_t)(k0 + 1) * cols + col] : 0.0f;
    dst[p] = (u32)f2bf(f0) | ((u32)f2bf(f1) << 16);
  }
}

// ---------------------------------------------------------------------------
// Generic tiled WMMA GEMM: C = ACT(A[M,Kd] @ B[Kd,Nd] + bias (+res))
// A bf16 row-major (2x b128/lane/step), B fragment-packed (1x 32B/lane/step).
// 4 waves/block, each wave one 16x16 tile; all flags compile-time.
// ---------------------------------------------------------------------------
template <int ACT, bool RESF, bool CFF, bool CBF>
__global__ __launch_bounds__(128)
void wmma_gemm_kernel(const u16* __restrict__ A, int lda,
                      const v8u* __restrict__ Bp,
                      const float* __restrict__ bias,
                      const float* __restrict__ res, int ldres,
                      float* __restrict__ Cf, u16* __restrict__ Cb, int ldc,
                      int Nd, int Kd) {
  int wave = threadIdx.x >> 5;
  int lane = threadIdx.x & 31;
  int tileM = blockIdx.y << 4;
  int tileN = (blockIdx.x * 4 + wave) << 4;
  if (tileN >= Nd) return;                       // uniform per wave
  int m  = lane & 15;
  int hh = lane >> 4;
  const u16* Arow = A + (size_t)(tileM + m) * lda + (hh << 3);
  const v8u* Brow = Bp + ((size_t)(tileN >> 4) * (Kd >> 5)) * 32 + lane;
  v8f acc = {};
  for (int kb = 0; kb < Kd; kb += 32) {
    Frag a, b;
    a.h.lo = *(const v4u*)(Arow + kb);           // K 0..7   (+8 per half-wave)
    a.h.hi = *(const v4u*)(Arow + kb + 16);      // K 16..23
    b.u = Brow[(size_t)(kb >> 5) * 32];
    acc = __builtin_amdgcn_wmma_f32_16x16x32_bf16(false, a.bf, false, b.bf,
                                                  (short)0, acc, false, false);
  }
  int col = tileN + (lane & 15);
  float bv = bias[col];
#pragma unroll
  for (int r = 0; r < 8; ++r) {
    int row = tileM + r + (hh << 3);
    float v = acc[r] + bv;
    if (RESF) v += res[(size_t)row * ldres + col];
    if (ACT == 1) v = silu_f(v);
    if (ACT == 2) v = fmaxf(v, 0.0f);
    if (CFF) Cf[(size_t)row * ldc + col] = v;
    if (CBF) Cb[(size_t)row * ldc + col] = f2bf(v);
  }
}

// ---------------------------------------------------------------------------
// Fused EGNN edge MLP: per atom i, build [16,544] edge-input tile in LDS,
// GEMM1 (544->1056, SiLU) streamed 32 hidden cols at a time through LDS,
// each strip immediately consumed as the K-slice of GEMM2 (1056->16, SiLU),
// then masked reduce over the 16 neighbors -> m_i[i][16].
// ---------------------------------------------------------------------------
__global__ __launch_bounds__(32)
void egnn_edge_kernel(const u16* __restrict__ featsB,
                      const int* __restrict__ knn_idx,
                      const float* __restrict__ knn_rd,
                      const v8u* __restrict__ W1p,   // fragment-packed [EK,EH]
                      const float* __restrict__ b1e, // [EH]
                      const v8u* __restrict__ W2p,   // fragment-packed [EH,16]
                      const float* __restrict__ b2e, // [16]
                      float* __restrict__ m_i) {     // [N, 16]
  __shared__ u16 sA[16 * EK];
  __shared__ u16 sH[16 * 32];
  __shared__ float sRed[32];
  int i = blockIdx.x;
  int lane = threadIdx.x;
  // Build edge_input tile: row kk = neighbor slot, cols [f_i | f_j | rd | pad]
  for (int t = lane; t < 16 * EK; t += 32) {
    int kk = t / EK, c = t % EK;
    u16 v;
    if (c < DIMC) v = featsB[(size_t)i * DIMC + c];
    else if (c < 2 * DIMC) {
      int j = knn_idx[i * KNNc + kk];
      v = featsB[(size_t)j * DIMC + (c - DIMC)];
    } else if (c == 2 * DIMC) v = f2bf(knn_rd[i * KNNc + kk]);
    else v = 0;
    sA[t] = v;
  }
  __syncthreads();
  int m = lane & 15, hh = lane >> 4;
  v8f acc2 = {};
#pragma unroll 1
  for (int t = 0; t < EH / 32; ++t) {
    for (int sub = 0; sub < 2; ++sub) {
      int tn = (t << 1) + sub;                   // hidden 16-col tile index
      v8f hacc = {};
      int col = (tn << 4) + (lane & 15);
      for (int kb = 0; kb < EK; kb += 32) {
        Frag a, b;
        a.h.lo = *(const v4u*)&sA[m * EK + kb + (hh << 3)];
        a.h.hi = *(const v4u*)&sA[m * EK + kb + 16 + (hh << 3)];
        b.u = W1p[((size_t)tn * (EK >> 5) + (kb >> 5)) * 32 + lane];
        hacc = __builtin_amdgcn_wmma_f32_16x16x32_bf16(false, a.bf, false, b.bf,
                                                       (short)0, hacc, false, false);
      }
#pragma unroll
      for (int r = 0; r < 8; ++r) {
        float v = silu_f(hacc[r] + b1e[col]);
        sH[(r + (hh << 3)) * 32 + (sub << 4) + (lane & 15)] = f2bf(v);
      }
    }
    __syncthreads();
    Frag a2, b2;
    a2.h.lo = *(const v4u*)&sH[m * 32 + (hh << 3)];
    a2.h.hi = *(const v4u*)&sH[m * 32 + 16 + (hh << 3)];
    b2.u = W2p[(size_t)t * 32 + lane];           // tn=0, kbi=t
    acc2 = __builtin_amdgcn_wmma_f32_16x16x32_bf16(false, a2.bf, false, b2.bf,
                                                   (short)0, acc2, false, false);
    __syncthreads();
  }
  // acc2: rows = neighbor kk = r + 8*hh, cols = m_dim channel = lane&15
  float part = 0.0f;
  int c = lane & 15;
#pragma unroll
  for (int r = 0; r < 8; ++r) {
    int kk = r + (hh << 3);
    float v = silu_f(acc2[r] + b2e[c]);
    float rd = knn_rd[i * KNNc + kk];
    part += (rd <= 5.0f) ? v : 0.0f;             // VALID_R vs squared dist
  }
  sRed[lane] = part;
  __syncthreads();
  if (lane < 16) m_i[i * KNNc + lane] = sRed[lane] + sRed[lane + 16];
}

// ---------------------------------------------------------------------------
struct EmbPtrs { const float* e[9]; };

__global__ __launch_bounds__(256)
void atom_encode_kernel(const int* __restrict__ x, EmbPtrs emb,
                        float* __restrict__ fF, u16* __restrict__ fB) {
  int i = blockIdx.x, c = threadIdx.x;
  const int* xi = x + i * 9;
  float s = 0.0f;
#pragma unroll
  for (int f = 0; f < 9; ++f) s += emb.e[f][(size_t)xi[f] * DIMC + c];
  fF[(size_t)i * DIMC + c] = s;
  fB[(size_t)i * DIMC + c] = f2bf(s);
}

__global__ __launch_bounds__(256)
void knn_kernel(const float* __restrict__ pos, int* __restrict__ knn_idx,
                float* __restrict__ knn_rd) {
  __shared__ float sv[256];
  __shared__ int   si[256];
  __shared__ int   sel[KNNc];
  int i = blockIdx.x, t = threadIdx.x;
  float px = pos[i * 3], py = pos[i * 3 + 1], pz = pos[i * 3 + 2];
  for (int it = 0; it < KNNc; ++it) {
    float best = 3.0e38f; int bidx = 0x7fffffff;
    for (int j = t; j < Nn; j += 256) {
      bool used = false;
      for (int u = 0; u < it; ++u) used |= (sel[u] == j);
      if (used) continue;
      float dx = px - pos[j * 3], dy = py - pos[j * 3 + 1], dz = pz - pos[j * 3 + 2];
      float d = dx * dx + dy * dy + dz * dz;
      if (d < best || (d == best && j < bidx)) { best = d; bidx = j; }
    }
    sv[t] = best; si[t] = bidx;
    __syncthreads();
    for (int s = 128; s > 0; s >>= 1) {
      if (t < s && (sv[t + s] < sv[t] || (sv[t + s] == sv[t] && si[t + s] < si[t]))) {
        sv[t] = sv[t + s]; si[t] = si[t + s];
      }
      __syncthreads();
    }
    if (t == 0) { sel[it] = si[0]; knn_idx[i * KNNc + it] = si[0]; knn_rd[i * KNNc + it] = sv[0]; }
    __syncthreads();
  }
}

__global__ __launch_bounds__(256)
void nodein_kernel(const float* __restrict__ fF, const float* __restrict__ m_i,
                   const float* __restrict__ g, const float* __restrict__ b,
                   u16* __restrict__ nin) {           // [N, 288]
  __shared__ float red[256];
  int i = blockIdx.x, t = threadIdx.x;
  float v = fF[(size_t)i * DIMC + t];
  red[t] = v; __syncthreads();
  for (int s = 128; s > 0; s >>= 1) { if (t < s) red[t] += red[t + s]; __syncthreads(); }
  float mu = red[0] * (1.0f / DIMC);
  __syncthreads();
  float d = v - mu;
  red[t] = d * d; __syncthreads();
  for (int s = 128; s > 0; s >>= 1) { if (t < s) red[t] += red[t + s]; __syncthreads(); }
  float var = red[0] * (1.0f / DIMC);
  float nv = d * rsqrtf(var + 1e-5f) * g[t] + b[t];
  nin[(size_t)i * 288 + t] = f2bf(nv);
  if (t < 32) nin[(size_t)i * 288 + 256 + t] = (t < 16) ? f2bf(m_i[i * KNNc + t]) : (u16)0;
}

__global__ __launch_bounds__(256)
void scatter_add_kernel(const float* __restrict__ src, const int* __restrict__ gidx,
                        const int* __restrict__ sidx, float* __restrict__ dst, int C) {
  int e = blockIdx.x, t = threadIdx.x;
  int srow = gidx ? gidx[e] : e;
  int drow = sidx[e];
  for (int c = t; c < C; c += 256)
    atomicAdd(&dst[(size_t)drow * C + c], src[(size_t)srow * C + c]);
}

__global__ __launch_bounds__(256)
void cat2_kernel(const u16* __restrict__ hB, const float* __restrict__ xe,
                 const int* __restrict__ V, const int* __restrict__ E,
                 u16* __restrict__ out) {            // [NNZ, 512]
  int e = blockIdx.x, t = threadIdx.x;
  int v = V[e], he = E[e];
  out[(size_t)e * 512 + t]       = hB[(size_t)v * DIMC + t];
  out[(size_t)e * 512 + 256 + t] = f2bf(xe[(size_t)he * DIMC + t]);
}

__global__ __launch_bounds__(256)
void cat3_kernel(const float* __restrict__ xv, const u16* __restrict__ x0B,
                 u16* __restrict__ out) {            // [N, 512]
  int i = blockIdx.x, t = threadIdx.x;
  out[(size_t)i * 512 + t]       = f2bf(xv[(size_t)i * DIMC + t]);
  out[(size_t)i * 512 + 256 + t] = x0B[(size_t)i * DIMC + t];
}

__global__ __launch_bounds__(256)
void pool_kernel(const u16* __restrict__ hB, const int* __restrict__ batch,
                 float* __restrict__ gF) {
  int i = blockIdx.x, t = threadIdx.x;
  atomicAdd(&gF[(size_t)batch[i] * DIMC + t], bf2f(hB[(size_t)i * DIMC + t]));
}

__global__ void f2b_kernel(const float* __restrict__ src, u16* __restrict__ dst, int n) {
  int i = blockIdx.x * 256 + threadIdx.x;
  if (i < n) dst[i] = f2bf(src[i]);
}

__global__ void pad_b_kernel(const float* __restrict__ src, float* __restrict__ dst,
                             int cols, int cp) {
  int c = blockIdx.x * 256 + threadIdx.x;
  if (c < cp) dst[c] = (c < cols) ? src[c] : 0.0f;
}

__global__ void extract_kernel(const float* __restrict__ src, float* __restrict__ out, int n) {
  int i = blockIdx.x * 256 + threadIdx.x;
  if (i < n) out[i] = src[(size_t)i * 16];
}

// ---------------------------------------------------------------------------
extern "C" void kernel_launch(void* const* d_in, const int* in_sizes, int n_in,
                              void* d_out, int out_size, void* d_ws, size_t ws_size,
                              hipStream_t stream) {
  (void)in_sizes; (void)n_in; (void)out_size; (void)ws_size;
  const int*   x     = (const int*)d_in[0];
  const float* pos   = (const float*)d_in[1];
  const int*   V     = (const int*)d_in[2];
  const int*   E     = (const int*)d_in[3];
  const int*   batch = (const int*)d_in[4];
  EmbPtrs emb;
  for (int f = 0; f < 9; ++f) emb.e[f] = (const float*)d_in[5 + f];
  const float* edge1w = (const float*)d_in[14]; const float* edge1b = (const float*)d_in[15];
  const float* edge2w = (const float*)d_in[16]; const float* edge2b = (const float*)d_in[17];
  // 18..21: coors1/coors2 (dead: coors_out unused downstream), 28: coors_scale
  const float* node1w = (const float*)d_in[22]; const float* node1b = (const float*)d_in[23];
  const float* node2w = (const float*)d_in[24]; const float* node2b = (const float*)d_in[25];
  const float* nng    = (const float*)d_in[26]; const float* nnb    = (const float*)d_in[27];
  const float* W1w = (const float*)d_in[29]; const float* W1b = (const float*)d_in[30];
  const float* W2w = (const float*)d_in[31]; const float* W2b = (const float*)d_in[32];
  const float* W3w = (const float*)d_in[33]; const float* W3b = (const float*)d_in[34];
  const float* o1w = (const float*)d_in[35]; const float* o1b = (const float*)d_in[36];
  const float* o2w = (const float*)d_in[37]; const float* o2b = (const float*)d_in[38];

  char* wsp = (char*)d_ws; size_t off = 0;
  auto alloc = [&](size_t bytes) -> void* {
    off = (off + 255) & ~(size_t)255;
    void* p = wsp + off; off += bytes; return p;
  };
  float* featsF = (float*)alloc((size_t)Nn * DIMC * 4);
  u16*   featsB = (u16*)  alloc((size_t)Nn * DIMC * 2);
  int*   kidx   = (int*)  alloc((size_t)Nn * KNNc * 4);
  float* krd    = (float*)alloc((size_t)Nn * KNNc * 4);
  float* m_i    = (float*)alloc((size_t)Nn * KNNc * 4);
  // fragment-packed weights (2 bytes/element at padded dims)
  v8u* e1p = (v8u*)alloc((size_t)EK * EH * 2);   float* e1bP = (float*)alloc(EH * 4);
  v8u* e2p = (v8u*)alloc((size_t)EH * 16 * 2);   float* e2bP = (float*)alloc(16 * 4);
  v8u* n1p = (v8u*)alloc((size_t)288 * 512 * 2); float* n1bP = (float*)alloc(512 * 4);
  v8u* n2p = (v8u*)alloc((size_t)512 * 256 * 2); float* n2bP = (float*)alloc(256 * 4);
  v8u* w1p = (v8u*)alloc((size_t)256 * 256 * 2); float* w1bP = (float*)alloc(256 * 4);
  v8u* w2p = (v8u*)alloc((size_t)512 * 256 * 2); float* w2bP = (float*)alloc(256 * 4);
  v8u* w3p = (v8u*)alloc((size_t)512 * 256 * 2); float* w3bP = (float*)alloc(256 * 4);
  v8u* o1p = (v8u*)alloc((size_t)256 * 256 * 2); float* o1bP = (float*)alloc(256 * 4);
  v8u* o2p = (v8u*)alloc((size_t)256 * 16 * 2);  float* o2bP = (float*)alloc(16 * 4);
  u16*   nin  = (u16*)alloc((size_t)Nn * 288 * 2);
  u16*   nh   = (u16*)alloc((size_t)Nn * 512 * 2);
  u16*   x0B  = (u16*)alloc((size_t)Nn * DIMC * 2);
  u16*   hB   = (u16*)alloc((size_t)Nn * DIMC * 2);
  float* w1F  = (float*)alloc((size_t)Nn * DIMC * 4);
  float* xe   = (float*)alloc((size_t)NEc * DIMC * 4);
  u16*   cat2 = (u16*)alloc((size_t)NNZc * 512 * 2);
  float* xevF = (float*)alloc((size_t)NNZc * DIMC * 4);
  float* xv   = (float*)alloc((size_t)Nn * DIMC * 4);
  u16*   cat3 = (u16*)alloc((size_t)Nn * 512 * 2);
  float* gF   = (float*)alloc((size_t)Bc * DIMC * 4);
  u16*   gB   = (u16*)alloc((size_t)Bc * DIMC * 2);
  u16*   ghB  = (u16*)alloc((size_t)Bc * DIMC * 2);
  float* o2o  = (float*)alloc((size_t)Bc * 16 * 4);

  // Weight prep: fragment-major bf16 pack + bias zero-pad
  auto pack = [&](const float* src, int rows, int cols, v8u* dst, int Kd, int Nd) {
    pack_w_kernel<<<(Nd / 16) * (Kd / 32), 32, 0, stream>>>(src, rows, cols, (u32*)dst, Kd, Nd);
  };
  pack(edge1w, 513, 1026, e1p, EK, EH);
  pack(edge2w, 1026, 16, e2p, EH, 16);
  pack(node1w, 272, 512, n1p, 288, 512);
  pack(node2w, 512, 256, n2p, 512, 256);
  pack(W1w, 256, 256, w1p, 256, 256);
  pack(W2w, 512, 256, w2p, 512, 256);
  pack(W3w, 512, 256, w3p, 512, 256);
  pack(o1w, 256, 256, o1p, 256, 256);
  pack(o2w, 256, 1, o2p, 256, 16);
  pad_b_kernel<<<(EH + 255) / 256, 256, 0, stream>>>(edge1b, e1bP, 1026, EH);
  pad_b_kernel<<<1, 256, 0, stream>>>(edge2b, e2bP, 16, 16);
  pad_b_kernel<<<2, 256, 0, stream>>>(node1b, n1bP, 512, 512);
  pad_b_kernel<<<1, 256, 0, stream>>>(node2b, n2bP, 256, 256);
  pad_b_kernel<<<1, 256, 0, stream>>>(W1b, w1bP, 256, 256);
  pad_b_kernel<<<1, 256, 0, stream>>>(W2b, w2bP, 256, 256);
  pad_b_kernel<<<1, 256, 0, stream>>>(W3b, w3bP, 256, 256);
  pad_b_kernel<<<1, 256, 0, stream>>>(o1b, o1bP, 256, 256);
  pad_b_kernel<<<1, 256, 0, stream>>>(o2b, o2bP, 1, 16);

  // Atom encoder + kNN + fused EGNN edge MLP (-> m_i)
  atom_encode_kernel<<<Nn, 256, 0, stream>>>(x, emb, featsF, featsB);
  knn_kernel<<<Nn, 256, 0, stream>>>(pos, kidx, krd);
  egnn_edge_kernel<<<Nn, 32, 0, stream>>>(featsB, kidx, krd, e1p, e1bP, e2p, e2bP, m_i);

  // Node MLP: layernorm(feats) ++ m_i -> 512 (silu) -> 256, + feats residual
  nodein_kernel<<<Nn, 256, 0, stream>>>(featsF, m_i, nng, nnb, nin);
  wmma_gemm_kernel<1, false, false, true><<<dim3(8, Nn / 16), 128, 0, stream>>>(
      nin, 288, n1p, n1bP, nullptr, 0, nullptr, nh, 512, 512, 288);
  wmma_gemm_kernel<0, true, false, true><<<dim3(4, Nn / 16), 128, 0, stream>>>(
      nh, 512, n2p, n2bP, featsF, 256, nullptr, x0B, 256, 256, 512);
  hipMemcpyAsync(hB, x0B, (size_t)Nn * DIMC * 2, hipMemcpyDeviceToDevice, stream);

  // 3x shared MHNNSConv: V->E->V gather/MLP/scatter
  for (int layer = 0; layer < 3; ++layer) {
    wmma_gemm_kernel<0, false, true, false><<<dim3(4, Nn / 16), 128, 0, stream>>>(
        hB, 256, w1p, w1bP, nullptr, 0, w1F, nullptr, 256, 256, 256);
    hipMemsetAsync(xe, 0, (size_t)NEc * DIMC * 4, stream);
    scatter_add_kernel<<<NNZc, 256, 0, stream>>>(w1F, V, E, xe, DIMC);
    cat2_kernel<<<NNZc, 256, 0, stream>>>(hB, xe, V, E, cat2);
    wmma_gemm_kernel<0, false, true, false><<<dim3(4, NNZc / 16), 128, 0, stream>>>(
        cat2, 512, w2p, w2bP, nullptr, 0, xevF, nullptr, 256, 256, 512);
    hipMemsetAsync(xv, 0, (size_t)Nn * DIMC * 4, stream);
    scatter_add_kernel<<<NNZc, 256, 0, stream>>>(xevF, nullptr, V, xv, DIMC);
    cat3_kernel<<<Nn, 256, 0, stream>>>(xv, x0B, cat3);
    wmma_gemm_kernel<2, false, false, true><<<dim3(4, Nn / 16), 128, 0, stream>>>(
        cat3, 512, w3p, w3bP, nullptr, 0, nullptr, hB, 256, 256, 512);
  }

  // Global add pool + output MLP
  hipMemsetAsync(gF, 0, (size_t)Bc * DIMC * 4, stream);
  pool_kernel<<<Nn, 256, 0, stream>>>(hB, batch, gF);
  f2b_kernel<<<(Bc * DIMC + 255) / 256, 256, 0, stream>>>(gF, gB, Bc * DIMC);
  wmma_gemm_kernel<2, false, false, true><<<dim3(4, Bc / 16), 128, 0, stream>>>(
      gB, 256, o1p, o1bP, nullptr, 0, nullptr, ghB, 256, 256, 256);
  wmma_gemm_kernel<0, false, true, false><<<dim3(1, Bc / 16), 128, 0, stream>>>(
      ghB, 256, o2p, o2bP, nullptr, 0, o2o, nullptr, 16, 16, 256);
  extract_kernel<<<(Bc + 255) / 256, 256, 0, stream>>>(o2o, (float*)d_out, Bc);
}
